// DoRALinear_64811056496880
// MI455X (gfx1250) — matrix operand order repaired and checked by reading, not compiled, not run on hardware
//
#include <hip/hip_runtime.h>
#include <math.h>

typedef _Float16 v8h  __attribute__((ext_vector_type(8)));
typedef _Float16 v16h __attribute__((ext_vector_type(16)));
typedef float    v8f  __attribute__((ext_vector_type(8)));

#define M_DIM 8192   // B*S tokens
#define K_DIM 4096   // in features
#define N_DIM 4096   // out features
#define RANK  16
#define SCALING_F 2.0f   // alpha/rank = 32/16

// ---------------- kernel 1: W'h[o,:] = (mag[o]/||W+2BA||) * (W[o,:]+2*(BA)[o,:])
// one 256-thread block per output row o; each thread owns 16 columns.
__global__ __launch_bounds__(256)
void k_wprime(const float* __restrict__ W,  const float* __restrict__ Aw,
              const float* __restrict__ Bw, const float* __restrict__ mag,
              _Float16* __restrict__ Wh) {
  const int o   = blockIdx.x;
  const int tid = threadIdx.x;
  __shared__ float bb[RANK];
  __shared__ float red[256];
  if (tid < RANK) bb[tid] = Bw[o * RANK + tid];
  __syncthreads();

  float adapted[16];
  float sumsq = 0.0f;
#pragma unroll
  for (int c = 0; c < 16; ++c) {
    const int i = c * 256 + tid;
    float bal = 0.0f;
#pragma unroll
    for (int r = 0; r < RANK; ++r)
      bal = fmaf(bb[r], Aw[(size_t)r * K_DIM + i], bal);
    const float ad = fmaf(SCALING_F, bal, W[(size_t)o * K_DIM + i]);
    adapted[c] = ad;
    sumsq = fmaf(ad, ad, sumsq);
  }
  red[tid] = sumsq;
  __syncthreads();
  for (int s = 128; s > 0; s >>= 1) {
    if (tid < s) red[tid] += red[tid + s];
    __syncthreads();
  }
  const float scale = mag[o] / sqrtf(red[0]);
#pragma unroll
  for (int c = 0; c < 16; ++c)
    Wh[(size_t)o * K_DIM + c * 256 + tid] = (_Float16)(scale * adapted[c]);
}

// ---------------- kernel 2: out[m,n] = sum_k x[m,k] * Wh[n,k]
// x loaded as f32 and converted to f16 in-register (fused cvt);
// f16 WMMA with f32 accumulation. 4 waves/block; each wave owns a
// 64x64 C block (4x4 tiles of 16x16).
__global__ __launch_bounds__(128)
void k_gemm(const float* __restrict__ x, const _Float16* __restrict__ wh,
            float* __restrict__ out) {
  const int lane  = threadIdx.x & 31;
  const int wv    = threadIdx.x >> 5;
  const int lid   = lane & 15;   // row/col within tile
  const int hi    = lane >> 4;   // lane-half selects K sub-chunk
  const int nBase = blockIdx.x * 64;
  const int mBase = blockIdx.y * 256 + wv * 64;

  // ISA 16-bit A layout: lane holds K=[k0+8*hi,+8) and [k0+16+8*hi,+8).
  // ISA 16-bit B layout: lane holds K=[k0+8*hi,+8) and [k0+16+8*hi,+8) of column n (a row of W').
  const float*    aRow[4];
  const _Float16* bRow[4];
#pragma unroll
  for (int t = 0; t < 4; ++t) {
    aRow[t] = x  + (size_t)(mBase + 16 * t + lid) * K_DIM + hi * 8;
    bRow[t] = wh + (size_t)(nBase + 16 * t + lid) * K_DIM + hi * 8;
  }

  v8f acc[4][4] = {};

  for (int k0 = 0; k0 < K_DIM; k0 += 32) {
    __builtin_prefetch(aRow[0] + k0 + 32, 0, 3);  // next K-chunk (f32: 128B ahead)
    __builtin_prefetch(bRow[0] + k0 + 64, 0, 3);
    v16h a[4], b[4];
#pragma unroll
    for (int t = 0; t < 4; ++t) {
      // A: 8+8 f32 values, convert to f16 in-register
      float4 c0 = *(const float4*)(aRow[t] + k0);
      float4 c1 = *(const float4*)(aRow[t] + k0 + 4);
      float4 c2 = *(const float4*)(aRow[t] + k0 + 16);
      float4 c3 = *(const float4*)(aRow[t] + k0 + 20);
      a[t] = (v16h){(_Float16)c0.x, (_Float16)c0.y, (_Float16)c0.z, (_Float16)c0.w,
                    (_Float16)c1.x, (_Float16)c1.y, (_Float16)c1.z, (_Float16)c1.w,
                    (_Float16)c2.x, (_Float16)c2.y, (_Float16)c2.z, (_Float16)c2.w,
                    (_Float16)c3.x, (_Float16)c3.y, (_Float16)c3.z, (_Float16)c3.w};
      // B: already f16, two contiguous b128 loads
      v8h blo = *(const v8h*)(bRow[t] + k0);
      v8h bhi = *(const v8h*)(bRow[t] + k0 + 16);
      b[t] = __builtin_shufflevector(blo, bhi,
              0,1,2,3,4,5,6,7,8,9,10,11,12,13,14,15);
    }
#pragma unroll
    for (int ti = 0; ti < 4; ++ti)
#pragma unroll
      for (int tj = 0; tj < 4; ++tj)
        acc[ti][tj] = __builtin_amdgcn_wmma_f32_16x16x32_f16(
            false, a[ti], false, b[tj], (short)0, acc[ti][tj], false, false);
  }

  // C layout: VGPR j, lane l -> M = j + 8*(l>=16), N = l%16
#pragma unroll
  for (int ti = 0; ti < 4; ++ti) {
    const int row0 = mBase + 16 * ti + hi * 8;
#pragma unroll
    for (int tj = 0; tj < 4; ++tj) {
      const int col = nBase + 16 * tj + lid;
#pragma unroll
      for (int j = 0; j < 8; ++j)
        out[(size_t)(row0 + j) * N_DIM + col] = acc[ti][tj][j];
    }
  }
}

extern "C" void kernel_launch(void* const* d_in, const int* in_sizes, int n_in,
                              void* d_out, int out_size, void* d_ws, size_t ws_size,
                              hipStream_t stream) {
  (void)in_sizes; (void)n_in; (void)out_size; (void)ws_size;
  const float* x   = (const float*)d_in[0];   // [8192, 4096]
  const float* W   = (const float*)d_in[1];   // [4096, 4096]
  const float* Aw  = (const float*)d_in[2];   // [16, 4096]
  const float* Bw  = (const float*)d_in[3];   // [4096, 16]
  const float* mag = (const float*)d_in[4];   // [4096]
  float* out = (float*)d_out;                 // [8192, 4096]

  // workspace: W'h (32 MB f16); x (128 MB f32) + W'h stay L2-resident in GEMM
  _Float16* wh = (_Float16*)d_ws;

  k_wprime<<<N_DIM, 256, 0, stream>>>(W, Aw, Bw, mag, wh);
  dim3 grid(N_DIM / 64, M_DIM / 256);
  k_gemm<<<grid, 128, 0, stream>>>(x, wh, out);
}
